// LoTNext_35570919145715
// MI455X (gfx1250) — matrix operand order repaired and checked by reading, not compiled, not run on hardware
//
#include <hip/hip_runtime.h>

// ---------------------------------------------------------------------------
// Shapes (compile-time constants from the reference)
//   B=32, L=100, H=250, D=256, NHEAD=8, hd=32, NHID=1024, P=10000, U=5000,
//   E1=E2=100000, N_nodes=U+P=15000 (padded 15008), K(fc)=500 (padded 512)
// ---------------------------------------------------------------------------

typedef __attribute__((ext_vector_type(16))) _Float16 v16h;
typedef __attribute__((ext_vector_type(8)))  _Float16 v8h;
typedef __attribute__((ext_vector_type(8)))  float    v8f;

// ---------------------------------------------------------------------------
// WMMA GEMM with 2x2 register blocking: each wave computes a 32x32 C tile
// using 4 accumulators -> 16 FLOP/byte of fragment traffic (vs 8 for 1 tile).
//   C[M x N] = act( A[M x K](f16) * Bt[N x K](f16)^T + bias )
//   A row-major (lda=K), Bt row-major (N rows of K) == B column-major.
//   M multiple of 32; grid.x covers ceil(N/32) with Bt padded to 32 rows;
//   K multiple of 32.  act: 0 none, 1 relu, 2 leaky(0.01).
//   outmode: 0 row-major ldc; 1 final remap row r=l*32+b -> C[(b*100+l)*ldc+n]
// ---------------------------------------------------------------------------
__global__ __launch_bounds__(32)
void wmma_gemm(const _Float16* __restrict__ A, const _Float16* __restrict__ Bt,
               const float* __restrict__ bias, float* __restrict__ C,
               int M, int N, int K, int ldc, int Nvalid, int act, int outmode)
{
  const int tn   = blockIdx.x * 32;
  const int tm   = blockIdx.y * 32;
  const int lane = threadIdx.x;
  const int l15  = lane & 15;
  const int lh   = lane >> 4;

  // A 16x32 layout: lane half selects K 0-15 vs 16-31 (matches B)
  const _Float16* Ap0 = A  + (size_t)(tm + l15) * K + lh * 16;
  const _Float16* Ap1 = Ap0 + (size_t)16 * K;
  // B: column per lane, 16 consecutive K per lane half
  const _Float16* Bp0 = Bt + (size_t)(tn + l15) * K + lh * 16;
  const _Float16* Bp1 = Bp0 + (size_t)16 * K;

  v8f acc00 = {}, acc01 = {}, acc10 = {}, acc11 = {};
  for (int kb = 0; kb < K; kb += 32) {
    v8h a0lo = *(const v8h*)(Ap0 + kb);
    v8h a0hi = *(const v8h*)(Ap0 + kb + 8);
    v8h a1lo = *(const v8h*)(Ap1 + kb);
    v8h a1hi = *(const v8h*)(Ap1 + kb + 8);
    v16h b0 = *(const v16h*)(Bp0 + kb);
    v16h b1 = *(const v16h*)(Bp1 + kb);
    v16h a0, a1;
#pragma unroll
    for (int i = 0; i < 8; ++i) {
      a0[i] = a0lo[i]; a0[8 + i] = a0hi[i];
      a1[i] = a1lo[i]; a1[8 + i] = a1hi[i];
    }
    acc00 = __builtin_amdgcn_wmma_f32_16x16x32_f16(false, a0, false, b0, (short)0, acc00, false, false);
    acc01 = __builtin_amdgcn_wmma_f32_16x16x32_f16(false, a0, false, b1, (short)0, acc01, false, false);
    acc10 = __builtin_amdgcn_wmma_f32_16x16x32_f16(false, a1, false, b0, (short)0, acc10, false, false);
    acc11 = __builtin_amdgcn_wmma_f32_16x16x32_f16(false, a1, false, b1, (short)0, acc11, false, false);
  }

  const int col0 = tn + l15;
  const int col1 = col0 + 16;
  const int rb   = tm + lh * 8;

  auto emit = [&](const v8f& acc, int rowbase, int col) {
    if (col >= Nvalid) return;
    const float bsc = bias ? bias[col] : 0.0f;
#pragma unroll
    for (int r = 0; r < 8; ++r) {
      int row = rowbase + r;
      float v = acc[r] + bsc;
      if (act == 1)      v = fmaxf(v, 0.0f);
      else if (act == 2) v = (v > 0.0f) ? v : 0.01f * v;
      size_t idx;
      if (outmode == 0) idx = (size_t)row * ldc + col;
      else { int l = row >> 5, b = row & 31; idx = ((size_t)b * 100 + l) * ldc + col; }
      C[idx] = v;
    }
  };
  emit(acc00, rb,      col0);
  emit(acc01, rb,      col1);
  emit(acc10, rb + 16, col0);
  emit(acc11, rb + 16, col1);
}

// ---- utility kernels ------------------------------------------------------

__global__ void k_zero(float* __restrict__ p, int n) {
  int i = blockIdx.x * blockDim.x + threadIdx.x;
  if (i < n) p[i] = 0.0f;
}

// W[K x N] f32  ->  out[Npad x Kpad] f16 (transposed, zero padded)
__global__ void k_wtrans(const float* __restrict__ W, _Float16* __restrict__ out,
                         int K, int N, int Kpad, int Npad) {
  int idx = blockIdx.x * blockDim.x + threadIdx.x;
  if (idx >= Npad * Kpad) return;
  int n = idx / Kpad, k = idx % Kpad;
  float v = (n < N && k < K) ? W[(size_t)k * N + n] : 0.0f;
  out[idx] = (_Float16)v;
}

// in[M x ldin] f32 -> out[Mpad x Kpad] f16 zero padded
__global__ void k_tohalf(const float* __restrict__ in, _Float16* __restrict__ out,
                         int M, int K, int ldin, int Kpad, int Mpad) {
  int idx = blockIdx.x * blockDim.x + threadIdx.x;
  if (idx >= Mpad * Kpad) return;
  int r = idx / Kpad, c = idx % Kpad;
  out[idx] = (_Float16)((r < M && c < K) ? in[(size_t)r * ldin + c] : 0.0f);
}

// ---- graph stages ---------------------------------------------------------

// encoder_weight[P x 250] += pg_val * enc_emb[pg_col]  (scatter to pg_row)
__global__ __launch_bounds__(256)
void k_scatter_pg(const int* __restrict__ row, const int* __restrict__ col,
                  const float* __restrict__ val, const float* __restrict__ emb,
                  float* __restrict__ encw) {
  int e = blockIdx.x, d = threadIdx.x;
  if (d >= 250) return;
  float v = val[e];
  atomicAdd(&encw[(size_t)row[e] * 250 + d], v * emb[(size_t)col[e] * 250 + d]);
}

// feats = [ew_user (U x 250) ; ew_poi0 (P x 250)]
__global__ __launch_bounds__(256)
void k_scatter_ig(const int* __restrict__ src, const int* __restrict__ dst,
                  const float* __restrict__ val, const float* __restrict__ emb,
                  const float* __restrict__ uemb, float* __restrict__ feats) {
  int e = blockIdx.x, d = threadIdx.x;
  if (d >= 250) return;
  float v = val[e];
  int s = src[e], t = dst[e];
  atomicAdd(&feats[(size_t)s * 250 + d],          v * emb[(size_t)t * 250 + d]);
  atomicAdd(&feats[(size_t)(5000 + t) * 250 + d], v * uemb[(size_t)s * 250 + d]);
}

// ew[i] = sigmoid( relu(e_in @ w1 + b1) @ w2 + b2 ),  e_in = [feats[src];feats[dst]]
__global__ __launch_bounds__(64)
void k_edge_mlp(const int* __restrict__ ig_src, const int* __restrict__ ig_dst,
                const float* __restrict__ feats, const float* __restrict__ w1,
                const float* __restrict__ b1, const float* __restrict__ w2,
                const float* __restrict__ b2, float* __restrict__ ew) {
  __shared__ float sh[500];
  __shared__ float red[64];
  int i = blockIdx.x, t = threadIdx.x;
  int s, d2;
  if (i < 100000) { s = ig_src[i];            d2 = ig_dst[i] + 5000; }
  else            { s = ig_dst[i - 100000] + 5000; d2 = ig_src[i - 100000]; }
  for (int k = t; k < 250; k += 64) {
    sh[k]       = feats[(size_t)s  * 250 + k];
    sh[250 + k] = feats[(size_t)d2 * 250 + k];
  }
  __syncthreads();
  float h = b1[t];
  for (int k = 0; k < 500; ++k) h = fmaf(sh[k], w1[k * 64 + t], h);
  h = fmaxf(h, 0.0f);
  red[t] = h * w2[t];
  __syncthreads();
  for (int st = 32; st > 0; st >>= 1) { if (t < st) red[t] += red[t + st]; __syncthreads(); }
  if (t == 0) ew[i] = 1.0f / (1.0f + expf(-(red[0] + b2[0])));
}

__global__ void k_gcn_den(const int* __restrict__ ig_src, const int* __restrict__ ig_dst,
                          const float* __restrict__ ew, float* __restrict__ den) {
  int i = blockIdx.x * blockDim.x + threadIdx.x;
  if (i >= 200000) return;
  int d = (i < 100000) ? (ig_dst[i] + 5000) : ig_src[i - 100000];
  atomicAdd(&den[d], ew[i]);
}

__global__ __launch_bounds__(256)
void k_gcn_num(const int* __restrict__ ig_src, const int* __restrict__ ig_dst,
               const float* __restrict__ ew, const float* __restrict__ h, int ldh,
               float* __restrict__ num) {
  int e = blockIdx.x, d = threadIdx.x;
  if (d >= 250) return;
  int s, t;
  if (e < 100000) { s = ig_src[e];            t = ig_dst[e] + 5000; }
  else            { s = ig_dst[e - 100000] + 5000; t = ig_src[e - 100000]; }
  atomicAdd(&num[(size_t)t * 250 + d], ew[e] * h[(size_t)s * ldh + d]);
}

// hin_h[15008 x 256] = f16( h + num/(den+1e-9) ), zero padded
__global__ void k_gcn_stage(const float* __restrict__ h, int ldh,
                            const float* __restrict__ num, const float* __restrict__ den,
                            _Float16* __restrict__ out) {
  int idx = blockIdx.x * blockDim.x + threadIdx.x;
  if (idx >= 15008 * 256) return;
  int r = idx >> 8, c = idx & 255;
  float v = 0.0f;
  if (r < 15000 && c < 250)
    v = h[(size_t)r * ldh + c] + num[(size_t)r * 250 + c] / (den[r] + 1e-9f);
  out[idx] = (_Float16)v;
}

// x_emb (b,l order, f16 into fused-input) + sim + time2vec columns
__global__ __launch_bounds__(256)
void k_xemb(const int* __restrict__ poi_ids, const int* __restrict__ time_slot,
            const int* __restrict__ user_id, const float* __restrict__ encw,
            const float* __restrict__ gcnout, const float* __restrict__ feats,
            const float* __restrict__ t2v_w0, const float* __restrict__ t2v_b0,
            const float* __restrict__ t2v_w, const float* __restrict__ t2v_b,
            float* __restrict__ sim, _Float16* __restrict__ fin) {
  int lb = blockIdx.x;                 // l*32 + b
  int l = lb >> 5, b = lb & 31;
  int t = threadIdx.x;
  __shared__ float red[256];
  int poi = poi_ids[b * 100 + l];
  int uid = user_id[b];
  size_t finrow = ((size_t)b * 100 + l) * 256;
  float sq = 0.0f;
  if (t < 250) {
    float xe = 0.5f * (encw[(size_t)poi * 250 + t] + gcnout[(size_t)(5000 + poi) * 256 + t]);
    fin[finrow + t] = (_Float16)xe;
    float diff = feats[(size_t)uid * 250 + t] - xe;   // ew_user[user_id]
    sq = diff * diff;
  } else {
    float tt = (float)time_slot[b * 100 + l] * (1.0f / 168.0f);
    float v = (t == 250) ? (tt * t2v_w0[0] + t2v_b0[0])
                         : sinf(tt * t2v_w[t - 251] + t2v_b[t - 251]);
    fin[finrow + t] = (_Float16)v;
  }
  red[t] = sq;
  __syncthreads();
  for (int st = 128; st > 0; st >>= 1) { if (t < st) red[t] += red[t + st]; __syncthreads(); }
  if (t == 0) sim[lb] = expf(-sqrtf(red[0]));
}

// attention scores + softmax: att[(b*8+h)*100+l][m]
__global__ __launch_bounds__(128)
void k_attn_score(const float* __restrict__ q, const float* __restrict__ k,
                  float* __restrict__ att) {
  int id = blockIdx.x;                 // b*800 + h*100 + l
  int l = id % 100, h = (id / 100) % 8, b = id / 800;
  int m = threadIdx.x;
  __shared__ float sred[128];
  float s = -1e30f;
  if (m < 100) {
    const float* qp = q + ((size_t)b * 100 + l) * 256 + h * 32;
    const float* kp = k + ((size_t)b * 100 + m) * 256 + h * 32;
    float acc = 0.0f;
#pragma unroll
    for (int d = 0; d < 32; ++d) acc = fmaf(qp[d], kp[d], acc);
    s = acc * 0.17677669529663687f;    // 1/sqrt(32)
  }
  sred[m] = s; __syncthreads();
  for (int st = 64; st > 0; st >>= 1) { if (m < st) sred[m] = fmaxf(sred[m], sred[m + st]); __syncthreads(); }
  float mx = sred[0]; __syncthreads();
  float e = (m < 100) ? expf(s - mx) : 0.0f;
  sred[m] = e; __syncthreads();
  for (int st = 64; st > 0; st >>= 1) { if (m < st) sred[m] += sred[m + st]; __syncthreads(); }
  if (m < 100) att[(size_t)id * 100 + m] = e / sred[0];
}

__global__ __launch_bounds__(256)
void k_attn_o(const float* __restrict__ att, const float* __restrict__ v,
              _Float16* __restrict__ oh) {
  int bl = blockIdx.x;                 // b*100 + l
  int b = bl / 100, l = bl % 100;
  int t = threadIdx.x, h = t >> 5;
  const float* ap = att + ((size_t)(b * 8 + h) * 100 + l) * 100;
  float acc = 0.0f;
  for (int m = 0; m < 100; ++m)
    acc = fmaf(ap[m], v[((size_t)b * 100 + m) * 256 + t], acc);
  oh[(size_t)bl * 256 + t] = (_Float16)acc;
}

// layernorm(a + r) with gamma/beta; writes f32 (optional) and f16
__global__ __launch_bounds__(256)
void k_ln(const float* __restrict__ a, const float* __restrict__ rsd,
          const float* __restrict__ g, const float* __restrict__ bta,
          float* __restrict__ outf, _Float16* __restrict__ outh) {
  int r = blockIdx.x, t = threadIdx.x;
  __shared__ float red[256];
  float x = a[(size_t)r * 256 + t] + rsd[(size_t)r * 256 + t];
  red[t] = x; __syncthreads();
  for (int st = 128; st > 0; st >>= 1) { if (t < st) red[t] += red[t + st]; __syncthreads(); }
  float mean = red[0] * (1.0f / 256.0f); __syncthreads();
  float dv = x - mean;
  red[t] = dv * dv; __syncthreads();
  for (int st = 128; st > 0; st >>= 1) { if (t < st) red[t] += red[t + st]; __syncthreads(); }
  float y = dv * rsqrtf(red[0] * (1.0f / 256.0f) + 1e-5f) * g[t] + bta[t];
  if (outf) outf[(size_t)r * 256 + t] = y;
  outh[(size_t)r * 256 + t] = (_Float16)y;
}

// haversine weighting + out_w + out_pu staging (f16, rows l*32+b, K padded 512)
__global__ __launch_bounds__(256)
void k_spatial(const float* __restrict__ latitude, const float* __restrict__ longitude,
               const float* __restrict__ sim, const float* __restrict__ decC,
               const int* __restrict__ user_id, const float* __restrict__ uemb,
               _Float16* __restrict__ pu) {
  int ib = blockIdx.x;                 // i*32 + b
  int i = ib >> 5, b = ib & 31;
  int t = threadIdx.x;
  __shared__ float w[112];
  __shared__ float red[256];
  const float DEG = 0.017453292519943295f;
  float lai = latitude[b * 100 + i] * DEG;
  float loi = longitude[b * 100 + i] * DEG;
  if (t < 112) {
    float wj = 0.0f;
    if (t <= i && t < 100) {
      float laj = latitude[b * 100 + t] * DEG;
      float loj = longitude[b * 100 + t] * DEG;
      float sdla = sinf((lai - laj) * 0.5f);
      float sdlo = sinf((loi - loj) * 0.5f);
      float aa = sdla * sdla + cosf(lai) * cosf(laj) * sdlo * sdlo;
      aa = fminf(fmaxf(aa, 0.0f), 1.0f);
      float dd = 2.0f * 6371.0f * asinf(sqrtf(aa));
      wj = (expf(-0.1f * dd) + 1e-10f) * sim[t * 32 + b];
    }
    w[t] = wj;
  }
  __syncthreads();
  red[t] = (t < 112) ? w[t] : 0.0f; __syncthreads();
  for (int st = 128; st > 0; st >>= 1) { if (t < st) red[t] += red[t + st]; __syncthreads(); }
  float inv = 1.0f / red[0];
  size_t prow = (size_t)ib * 512;
  int uid = user_id[b];
  for (int c = t; c < 512; c += 256) {
    float val;
    if (c < 250) {
      float acc = 0.0f;
      for (int j = 0; j <= i; ++j)
        acc = fmaf(w[j], decC[((size_t)b * 100 + j) * 256 + c], acc);
      val = acc * inv;
    } else if (c < 500) {
      val = uemb[(size_t)uid * 250 + (c - 250)];
    } else val = 0.0f;
    pu[prow + c] = (_Float16)val;
  }
}

// ---------------------------------------------------------------------------

extern "C" void kernel_launch(void* const* d_in, const int* in_sizes, int n_in,
                              void* d_out, int out_size, void* d_ws, size_t ws_size,
                              hipStream_t stream) {
  (void)in_sizes; (void)n_in; (void)out_size; (void)ws_size;
  const int*   poi_ids   = (const int*)  d_in[0];
  const int*   time_slot = (const int*)  d_in[1];
  const int*   user_id   = (const int*)  d_in[2];
  const int*   pg_row    = (const int*)  d_in[3];
  const int*   pg_col    = (const int*)  d_in[4];
  const int*   ig_src    = (const int*)  d_in[5];
  const int*   ig_dst    = (const int*)  d_in[6];
  const float* latitude  = (const float*)d_in[7];
  const float* longitude = (const float*)d_in[8];
  const float* pg_val    = (const float*)d_in[9];
  const float* ig_val    = (const float*)d_in[10];
  const float* enc_emb   = (const float*)d_in[11];
  const float* uemb      = (const float*)d_in[12];
  const float* mlp_w1    = (const float*)d_in[13];
  const float* mlp_b1    = (const float*)d_in[14];
  const float* mlp_w2    = (const float*)d_in[15];
  const float* mlp_b2    = (const float*)d_in[16];
  const float* gcn_w1    = (const float*)d_in[17];
  const float* gcn_w2    = (const float*)d_in[18];
  const float* t2v_w0    = (const float*)d_in[19];
  const float* t2v_b0    = (const float*)d_in[20];
  const float* t2v_w     = (const float*)d_in[21];
  const float* t2v_b     = (const float*)d_in[22];
  const float* fuse_w    = (const float*)d_in[23];
  const float* fuse_b    = (const float*)d_in[24];
  const float* wq = (const float*)d_in[25]; const float* bq = (const float*)d_in[26];
  const float* wk = (const float*)d_in[27]; const float* bk = (const float*)d_in[28];
  const float* wv = (const float*)d_in[29]; const float* bv = (const float*)d_in[30];
  const float* wo = (const float*)d_in[31]; const float* bo = (const float*)d_in[32];
  const float* ln1_g = (const float*)d_in[33]; const float* ln1_b = (const float*)d_in[34];
  const float* ln2_g = (const float*)d_in[35]; const float* ln2_b = (const float*)d_in[36];
  const float* ff_w1 = (const float*)d_in[37]; const float* ff_b1 = (const float*)d_in[38];
  const float* ff_w2 = (const float*)d_in[39]; const float* ff_b2 = (const float*)d_in[40];
  const float* dec_w = (const float*)d_in[41]; const float* dec_b = (const float*)d_in[42];
  const float* fc_w  = (const float*)d_in[43]; const float* fc_b  = (const float*)d_in[44];

  // ---- workspace (d_ws): f16 weights + small persistent buffers (~17 MB) ---
  char* wsb = (char*)d_ws; size_t woff = 0;
  auto aws = [&](size_t bytes) -> char* {
    woff = (woff + 255) & ~(size_t)255; char* p = wsb + woff; woff += bytes; return p;
  };
  _Float16* fc_wT = (_Float16*)aws((size_t)10016 * 512 * 2);  // N padded to 32
  _Float16* gw1T  = (_Float16*)aws(256 * 256 * 2);
  _Float16* gw2T  = (_Float16*)aws(256 * 256 * 2);
  _Float16* fuseT = (_Float16*)aws(256 * 256 * 2);
  _Float16* wqT   = (_Float16*)aws(256 * 256 * 2);
  _Float16* wkT   = (_Float16*)aws(256 * 256 * 2);
  _Float16* wvT   = (_Float16*)aws(256 * 256 * 2);
  _Float16* woT   = (_Float16*)aws(256 * 256 * 2);
  _Float16* fw1T  = (_Float16*)aws(1024 * 256 * 2);
  _Float16* fw2T  = (_Float16*)aws(256 * 1024 * 2);
  _Float16* decT  = (_Float16*)aws(256 * 256 * 2);
  float*    sim   = (float*)   aws(3200 * 4);
  _Float16* fin_h = (_Float16*)aws(3200 * 256 * 2);
  _Float16* pu_h  = (_Float16*)aws(3200 * 512 * 2);

  // ---- scratch arenas inside d_out (fully overwritten by final fc GEMM) ---
  char* ob = (char*)d_out; size_t ooff = 0;
  auto aout = [&](size_t bytes) -> char* {
    ooff = (ooff + 255) & ~(size_t)255; char* p = ob + ooff; ooff += bytes; return p;
  };
  // phase 1 (graph / GCN), dead after k_xemb:
  float*    encw   = (float*)   aout((size_t)10000 * 250 * 4);
  float*    feats  = (float*)   aout((size_t)15000 * 250 * 4);
  float*    ew     = (float*)   aout((size_t)200000 * 4);
  float*    den    = (float*)   aout((size_t)15000 * 4);
  float*    num    = (float*)   aout((size_t)15000 * 250 * 4);
  float*    h1gcn  = (float*)   aout((size_t)15008 * 256 * 4);
  float*    gcnout = (float*)   aout((size_t)15008 * 256 * 4);
  _Float16* hin_h  = (_Float16*)aout((size_t)15008 * 256 * 2);
  // phase 2 (transformer), aliases phase 1:
  ooff = 0;
  float*    fusedf = (float*)   aout((size_t)3200 * 256 * 4);
  _Float16* fusedh = (_Float16*)aout((size_t)3200 * 256 * 2);
  float*    qf     = (float*)   aout((size_t)3200 * 256 * 4);
  float*    kf     = (float*)   aout((size_t)3200 * 256 * 4);
  float*    vf     = (float*)   aout((size_t)3200 * 256 * 4);
  float*    att    = (float*)   aout((size_t)32 * 8 * 100 * 100 * 4);
  _Float16* oh     = (_Float16*)aout((size_t)3200 * 256 * 2);
  float*    attnf  = (float*)   aout((size_t)3200 * 256 * 4);
  float*    h1f    = (float*)   aout((size_t)3200 * 256 * 4);
  _Float16* h1h    = (_Float16*)aout((size_t)3200 * 256 * 2);
  float*    ffh    = (float*)   aout((size_t)3200 * 1024 * 4);
  _Float16* ffhh   = (_Float16*)aout((size_t)3200 * 1024 * 2);
  float*    ffo    = (float*)   aout((size_t)3200 * 256 * 4);
  _Float16* ench   = (_Float16*)aout((size_t)3200 * 256 * 2);
  float*    decC   = (float*)   aout((size_t)3200 * 256 * 4);

  auto g1 = [](int n) { return dim3((n + 255) / 256); };

  // ---- stage 0: weight f16 transposes -------------------------------------
  k_wtrans<<<g1(256 * 256),   256, 0, stream>>>(gcn_w1, gw1T, 250, 250, 256, 256);
  k_wtrans<<<g1(256 * 256),   256, 0, stream>>>(gcn_w2, gw2T, 250, 250, 256, 256);
  k_wtrans<<<g1(256 * 256),   256, 0, stream>>>(fuse_w, fuseT, 256, 256, 256, 256);
  k_wtrans<<<g1(256 * 256),   256, 0, stream>>>(wq, wqT, 256, 256, 256, 256);
  k_wtrans<<<g1(256 * 256),   256, 0, stream>>>(wk, wkT, 256, 256, 256, 256);
  k_wtrans<<<g1(256 * 256),   256, 0, stream>>>(wv, wvT, 256, 256, 256, 256);
  k_wtrans<<<g1(256 * 256),   256, 0, stream>>>(wo, woT, 256, 256, 256, 256);
  k_wtrans<<<g1(1024 * 256),  256, 0, stream>>>(ff_w1, fw1T, 256, 1024, 256, 1024);
  k_wtrans<<<g1(1024 * 256),  256, 0, stream>>>(ff_w2, fw2T, 1024, 256, 1024, 256);
  k_wtrans<<<g1(256 * 256),   256, 0, stream>>>(dec_w, decT, 256, 250, 256, 256);
  k_wtrans<<<g1(10016 * 512), 256, 0, stream>>>(fc_w, fc_wT, 500, 10000, 512, 10016);

  // ---- stage 1-2: segment sums --------------------------------------------
  k_zero<<<g1(10000 * 250), 256, 0, stream>>>(encw, 10000 * 250);
  k_zero<<<g1(15000 * 250), 256, 0, stream>>>(feats, 15000 * 250);
  k_zero<<<g1(15000),       256, 0, stream>>>(den, 15000);
  k_scatter_pg<<<100000, 256, 0, stream>>>(pg_row, pg_col, pg_val, enc_emb, encw);
  k_scatter_ig<<<100000, 256, 0, stream>>>(ig_src, ig_dst, ig_val, enc_emb, uemb, feats);

  // ---- stage 3: edge MLP ---------------------------------------------------
  k_edge_mlp<<<200000, 64, 0, stream>>>(ig_src, ig_dst, feats, mlp_w1, mlp_b1,
                                        mlp_w2, mlp_b2, ew);
  k_gcn_den<<<g1(200000), 256, 0, stream>>>(ig_src, ig_dst, ew, den);

  // ---- stage 4: GCN layer 1 (WMMA) ----------------------------------------
  k_zero<<<g1(15000 * 250), 256, 0, stream>>>(num, 15000 * 250);
  k_gcn_num<<<200000, 256, 0, stream>>>(ig_src, ig_dst, ew, feats, 250, num);
  k_gcn_stage<<<g1(15008 * 256), 256, 0, stream>>>(feats, 250, num, den, hin_h);
  wmma_gemm<<<dim3(8, 469), 32, 0, stream>>>(hin_h, gw1T, nullptr, h1gcn,
                                             15008, 256, 256, 256, 250, 1, 0);
  // ---- GCN layer 2 (WMMA) --------------------------------------------------
  k_zero<<<g1(15000 * 250), 256, 0, stream>>>(num, 15000 * 250);
  k_gcn_num<<<200000, 256, 0, stream>>>(ig_src, ig_dst, ew, h1gcn, 256, num);
  k_gcn_stage<<<g1(15008 * 256), 256, 0, stream>>>(h1gcn, 256, num, den, hin_h);
  wmma_gemm<<<dim3(8, 469), 32, 0, stream>>>(hin_h, gw2T, nullptr, gcnout,
                                             15008, 256, 256, 256, 250, 0, 0);

  // ---- stage 5: x_emb / sim / fused input ---------------------------------
  k_xemb<<<3200, 256, 0, stream>>>(poi_ids, time_slot, user_id, encw, gcnout,
                                   feats, t2v_w0, t2v_b0, t2v_w, t2v_b, sim, fin_h);

  // ---- stage 6: fuse (WMMA, leaky relu) -----------------------------------
  wmma_gemm<<<dim3(8, 100), 32, 0, stream>>>(fin_h, fuseT, fuse_b, fusedf,
                                             3200, 256, 256, 256, 256, 2, 0);
  k_tohalf<<<g1(3200 * 256), 256, 0, stream>>>(fusedf, fusedh, 3200, 256, 256, 256, 3200);

  // ---- stage 7: q/k/v projections (WMMA) ----------------------------------
  wmma_gemm<<<dim3(8, 100), 32, 0, stream>>>(fusedh, wqT, bq, qf, 3200, 256, 256, 256, 256, 0, 0);
  wmma_gemm<<<dim3(8, 100), 32, 0, stream>>>(fusedh, wkT, bk, kf, 3200, 256, 256, 256, 256, 0, 0);
  wmma_gemm<<<dim3(8, 100), 32, 0, stream>>>(fusedh, wvT, bv, vf, 3200, 256, 256, 256, 256, 0, 0);

  // ---- stage 8-9: attention -----------------------------------------------
  k_attn_score<<<25600, 128, 0, stream>>>(qf, kf, att);
  k_attn_o<<<3200, 256, 0, stream>>>(att, vf, oh);
  wmma_gemm<<<dim3(8, 100), 32, 0, stream>>>(oh, woT, bo, attnf, 3200, 256, 256, 256, 256, 0, 0);

  // ---- stage 10: LN1 -------------------------------------------------------
  k_ln<<<3200, 256, 0, stream>>>(fusedf, attnf, ln1_g, ln1_b, h1f, h1h);

  // ---- stage 11: FFN (WMMA) + LN2 -----------------------------------------
  wmma_gemm<<<dim3(32, 100), 32, 0, stream>>>(h1h, fw1T, ff_b1, ffh, 3200, 1024, 256, 1024, 1024, 1, 0);
  k_tohalf<<<g1(3200 * 1024), 256, 0, stream>>>(ffh, ffhh, 3200, 1024, 1024, 1024, 3200);
  wmma_gemm<<<dim3(8, 100), 32, 0, stream>>>(ffhh, fw2T, ff_b2, ffo, 3200, 256, 1024, 256, 256, 0, 0);
  k_ln<<<3200, 256, 0, stream>>>(h1f, ffo, ln2_g, ln2_b, nullptr, ench);

  // ---- stage 12: decoder projection (WMMA) --------------------------------
  wmma_gemm<<<dim3(8, 100), 32, 0, stream>>>(ench, decT, dec_b, decC, 3200, 256, 256, 256, 250, 0, 0);

  // ---- stage 13: haversine weighting + out_pu -----------------------------
  k_spatial<<<3200, 256, 0, stream>>>(latitude, longitude, sim, decC, user_id, uemb, pu_h);

  // ---- stage 14: final fc GEMM (WMMA) writes d_out in (B,L,P) layout ------
  wmma_gemm<<<dim3(313, 100), 32, 0, stream>>>(pu_h, fc_wT, fc_b, (float*)d_out,
                                               3200, 10016, 512, 10000, 10000, 0, 1);
}
